// S4Model_3350074491552
// MI455X (gfx1250) — matrix operand order repaired and checked by reading, not compiled, not run on hardware
//
#include <hip/hip_runtime.h>

typedef __attribute__((ext_vector_type(16))) _Float16 v16h;
typedef __attribute__((ext_vector_type(8)))  _Float16 v8h;
typedef __attribute__((ext_vector_type(8)))  float    v8f;

constexpr int B_ = 32, L_ = 4096, H_ = 256, N2_ = 32, NL_ = 4, DOUT_ = 8;
constexpr int O2_ = 2 * H_; // 512

__device__ __forceinline__ float gelu_tanh(float y) {
    float t = 0.7978845608028654f * (y + 0.044715f * y * y * y);
    return 0.5f * y * (1.0f + tanhf(t));
}
__device__ __forceinline__ float sigmoidf_(float x) { return 1.0f / (1.0f + __expf(-x)); }

// ---------------- encoder: h[b,ch,l] = x[b,l]*ew[ch] + eb[ch] ----------------
__global__ void __launch_bounds__(256) encode_kernel(const float* __restrict__ x,
                                                     const float* __restrict__ ew,
                                                     const float* __restrict__ eb,
                                                     float* __restrict__ h) {
    size_t i = (size_t)blockIdx.x * blockDim.x + threadIdx.x;
    if (i >= (size_t)B_ * H_ * L_) return;
    int l  = (int)(i % L_);
    int ch = (int)((i / L_) % H_);
    int b  = (int)(i / ((size_t)H_ * L_));
    h[i] = x[(size_t)b * L_ + l] * ew[ch] + eb[ch];
}

// ---------------- weights fp32 -> f16 ----------------
__global__ void __launch_bounds__(256) cvt_w_kernel(const float* __restrict__ w,
                                                    _Float16* __restrict__ w16) {
    int i = blockIdx.x * blockDim.x + threadIdx.x;
    if (i < NL_ * O2_ * H_) w16[i] = (_Float16)w[i];
}

// ---------------- per-layer discretized coefficients ----------------
__global__ void __launch_bounds__(256) coeff_kernel(const float* __restrict__ log_dt,
                                                    const float* __restrict__ C_re,
                                                    const float* __restrict__ C_im,
                                                    const float* __restrict__ lAr,
                                                    const float* __restrict__ Aim,
                                                    int layer,
                                                    float* __restrict__ wr, float* __restrict__ wi,
                                                    float* __restrict__ cr, float* __restrict__ ci) {
    int i = blockIdx.x * blockDim.x + threadIdx.x; // over H*N2
    if (i >= H_ * N2_) return;
    int h = i / N2_;
    float dt = __expf(log_dt[layer * H_ + h]);
    int pi = layer * H_ * N2_ + i;
    float Ar = -__expf(lAr[pi]);
    float Ai = Aim[pi];
    float dr = Ar * dt, di = Ai * dt;
    float er = __expf(dr);
    float wre = er * __cosf(di);
    float wim = er * __sinf(di);
    // (w - 1) / A
    float nr = wre - 1.0f, ni = wim;
    float den = Ar * Ar + Ai * Ai;
    float fr = (nr * Ar + ni * Ai) / den;
    float fi = (ni * Ar - nr * Ai) / den;
    float cre = C_re[pi], cim = C_im[pi];
    wr[i] = wre; wi[i] = wim;
    cr[i] = cre * fr - cim * fi;
    ci[i] = cre * fi + cim * fr;
}

// ---------------- SSM scan: one wave per (b,h), one complex state per lane ----------------
__global__ void __launch_bounds__(256) ssm_scan_kernel(const float* __restrict__ U,
                                                       const float* __restrict__ wr,
                                                       const float* __restrict__ wi,
                                                       const float* __restrict__ cr,
                                                       const float* __restrict__ ci,
                                                       const float* __restrict__ Dp,
                                                       int layer,
                                                       _Float16* __restrict__ Y) {
    const int lane = threadIdx.x & 31;
    const int seq  = blockIdx.x * (blockDim.x >> 5) + (threadIdx.x >> 5); // b*H + h
    if (seq >= B_ * H_) return;
    const int h = seq & (H_ - 1);
    const size_t base = (size_t)seq * L_;
    const int idx = h * N2_ + lane;
    const float Wr = wr[idx], Wi = wi[idx], Cr = cr[idx], Ci = ci[idx];
    const float d = Dp[layer * H_ + h];
    float sr = 0.f, si = 0.f;
    for (int l = 0; l < L_; ++l) {
        if ((l & 63) == 0) __builtin_prefetch(U + base + l + 256, 0, 0);
        float u  = U[base + l];
        float tr = fmaf(sr, Wr, fmaf(-si, Wi, u));
        float ti = fmaf(sr, Wi, si * Wr);
        sr = tr; si = ti;
        float p = fmaf(Cr, sr, -Ci * si);
        p += __shfl_xor(p, 16, 32);
        p += __shfl_xor(p, 8, 32);
        p += __shfl_xor(p, 4, 32);
        p += __shfl_xor(p, 2, 32);
        p += __shfl_xor(p, 1, 32);
        if (lane == 0) {
            float y = 2.0f * p + d * u;
            Y[base + l] = (_Float16)gelu_tanh(y);
        }
    }
}

// ---------------- output linear (WMMA) + GLU + residual ----------------
// Block tile: M=128 (64 'a' rows + 64 'g' rows), N=64 cols, K=256 fully unrolled in chunks of 32.
// 8 waves: 4 along M x 2 along N; 4 wmma tiles per wave per chunk => 32 static v_wmma.
// All global staging reads are 16B per lane; all fragment reads are ds_load_b128.
__global__ void __launch_bounds__(256) glu_gemm_kernel(const _Float16* __restrict__ Yf,
                                                       const _Float16* __restrict__ Wt,   // layer base (512,256)
                                                       const float* __restrict__ bias,    // layer base (512)
                                                       const float* __restrict__ Hprev,   // (B,H,L)
                                                       float* __restrict__ preLN) {       // (B*L,H)
    __shared__ _Float16 As[128][40];   // [row][k]  row stride 80B (16B aligned)
    __shared__ _Float16 Bt[64][40];    // [col][k]  K-contiguous per column
    __shared__ float    Zs[128][65];

    const int tid  = threadIdx.x;
    const int lane = tid & 31, wid = tid >> 5;
    const int wm = wid & 3, wn = wid >> 2;
    const int col0  = blockIdx.x * 64;        // global column = b*L + l
    const int b     = col0 >> 12;             // L = 4096
    const int l0    = col0 & (L_ - 1);
    const int strip = blockIdx.y;             // 0..3, 64 channels each
    const int ch0   = strip * 64;

    const size_t ybase = (size_t)b * H_ * L_ + l0;

    // staging assignments (fixed per thread)
    const int ar   = tid >> 1;                 // A row 0..127
    const int aseg = (tid & 1) * 16;           // A k-segment 0 / 16
    const int am   = (ar < 64) ? (ch0 + ar) : (H_ + ch0 + (ar - 64));
    const int bk   = tid >> 3;                 // B k row 0..31
    const int bseg = (tid & 7) * 8;            // B col segment (8 cols)

    v8f acc[2][2] = {};

#pragma unroll
    for (int kc = 0; kc < H_; kc += 32) {
        __syncthreads();
        // stage A tile (128 x 32): two 16B vector copies per thread
        {
            const v8h* srcA = (const v8h*)(Wt + am * H_ + kc + aseg);
            *(v8h*)&As[ar][aseg]     = srcA[0];
            *(v8h*)&As[ar][aseg + 8] = srcA[1];
        }
        // stage B tile (32 K x 64 cols): one 16B global read, 8-way LDS transpose scatter
        {
            v8h bv = *(const v8h*)(Yf + ybase + (size_t)(kc + bk) * L_ + bseg);
#pragma unroll
            for (int j = 0; j < 8; ++j) Bt[bseg + j][bk] = bv[j];
        }
        __syncthreads();

        const int arow  = lane & 15;
        const int khalf = (lane >> 4) * 16;  // A: lanes 16-31 hold the K+16 group
        const int kbb   = (lane >> 4) * 16;  // B: lanes 16-31 hold the K+16 group
        v16h afrag[2], bfrag[2];
#pragma unroll
        for (int mi = 0; mi < 2; ++mi) {
            int rr = wm * 32 + mi * 16 + arow;
#pragma unroll
            for (int v = 0; v < 8; ++v) {
                int kk = khalf + 2 * v;
                afrag[mi][2 * v]     = As[rr][kk];
                afrag[mi][2 * v + 1] = As[rr][kk + 1];
            }
        }
        const int bcol = lane & 15;
#pragma unroll
        for (int ni = 0; ni < 2; ++ni) {
            int cc = wn * 32 + ni * 16 + bcol;
#pragma unroll
            for (int v = 0; v < 8; ++v) {
                int kk = kbb + 2 * v;      // 16 contiguous halves -> two b128 ds loads
                bfrag[ni][2 * v]     = Bt[cc][kk];
                bfrag[ni][2 * v + 1] = Bt[cc][kk + 1];
            }
        }
#pragma unroll
        for (int mi = 0; mi < 2; ++mi)
#pragma unroll
            for (int ni = 0; ni < 2; ++ni)
                acc[mi][ni] = __builtin_amdgcn_wmma_f32_16x16x32_f16(
                    false, afrag[mi], false, bfrag[ni], (short)0, acc[mi][ni], false, false);
    }

    __syncthreads();
    // spill accumulators (C/D layout: VGPR r -> M = r + (lane>=16 ? 8 : 0), N = lane&15)
    {
        const int nn = lane & 15;
        const int mb = (lane >> 4) * 8;
#pragma unroll
        for (int mi = 0; mi < 2; ++mi)
#pragma unroll
            for (int ni = 0; ni < 2; ++ni)
#pragma unroll
                for (int r = 0; r < 8; ++r)
                    Zs[wm * 32 + mi * 16 + mb + r][wn * 32 + ni * 16 + nn] = acc[mi][ni][r];
    }
    __syncthreads();

    // GLU + residual -> preLN (B*L, H). Consecutive lanes -> consecutive ch: coalesced stores.
    for (int i = tid; i < 64 * 64; i += 256) {
        int c = i >> 6, r = i & 63;
        int ch = ch0 + r;
        float a = Zs[r][c]      + bias[ch];
        float g = Zs[r + 64][c] + bias[H_ + ch];
        float val = a * sigmoidf_(g) +
                    Hprev[(size_t)b * H_ * L_ + (size_t)ch * L_ + l0 + c];
        preLN[((size_t)b * L_ + l0 + c) * H_ + ch] = val;
    }
}

// ---------------- LayerNorm over H, writing back to (B,H,L) ----------------
__global__ void __launch_bounds__(256) layernorm_kernel(const float* __restrict__ preLN,
                                                        const float* __restrict__ lw,
                                                        const float* __restrict__ lb,
                                                        float* __restrict__ Hout) {
    __shared__ float s1[256], s2[256];
    const int row = blockIdx.x;          // b*L + l
    const int ch  = threadIdx.x;
    float v = preLN[(size_t)row * H_ + ch];
    s1[ch] = v; s2[ch] = v * v;
    __syncthreads();
    for (int off = 128; off > 0; off >>= 1) {
        if (ch < off) { s1[ch] += s1[ch + off]; s2[ch] += s2[ch + off]; }
        __syncthreads();
    }
    float mu   = s1[0] * (1.0f / H_);
    float var  = s2[0] * (1.0f / H_) - mu * mu;
    float rstd = rsqrtf(var + 1e-5f);
    int b = row >> 12, l = row & (L_ - 1);
    Hout[(size_t)b * H_ * L_ + (size_t)ch * L_ + l] = (v - mu) * rstd * lw[ch] + lb[ch];
}

// ---------------- mean pool over L (async-LDS staged) ----------------
__global__ void __launch_bounds__(256) pool_kernel(const float* __restrict__ Hin,
                                                   float* __restrict__ pooled) {
    __shared__ float s[256];
    const int seq = blockIdx.x;  // b*H + ch
    float acc = 0.f;
    // Stage each 256-element chunk through LDS with async global->LDS copies
    // (GLOBAL_LOAD_ASYNC_TO_LDS_B32, GV mode: per-lane LDS dest + per-lane 64-bit address),
    // then consume after s_wait_asynccnt 0. Each lane owns its slot: no cross-lane hazard.
    unsigned lds_off = (unsigned)(size_t)&s[threadIdx.x];
    for (int c = 0; c < L_ / 256; ++c) {
        const float* src = Hin + (size_t)seq * L_ + c * 256 + threadIdx.x;
        asm volatile("global_load_async_to_lds_b32 %0, %1, off"
                     :: "v"(lds_off), "v"(src) : "memory");
        asm volatile("s_wait_asynccnt 0x0" ::: "memory");
        acc += s[threadIdx.x];
    }
    s[threadIdx.x] = acc;
    __syncthreads();
    for (int off = 128; off > 0; off >>= 1) {
        if (threadIdx.x < off) s[threadIdx.x] += s[threadIdx.x + off];
        __syncthreads();
    }
    if (threadIdx.x == 0) pooled[seq] = s[0] * (1.0f / L_);
}

// ---------------- decoder + sigmoid ----------------
__global__ void __launch_bounds__(256) decode_kernel(const float* __restrict__ pooled,
                                                     const float* __restrict__ dw,
                                                     const float* __restrict__ db,
                                                     float* __restrict__ out) {
    int t = blockIdx.x * blockDim.x + threadIdx.x;
    if (t >= B_ * DOUT_) return;
    int b = t >> 3, d = t & 7;
    float acc = db[d];
    for (int ch = 0; ch < H_; ++ch) acc += pooled[b * H_ + ch] * dw[d * H_ + ch];
    out[t] = sigmoidf_(acc);
}

extern "C" void kernel_launch(void* const* d_in, const int* in_sizes, int n_in,
                              void* d_out, int out_size, void* d_ws, size_t ws_size,
                              hipStream_t stream) {
    (void)in_sizes; (void)n_in; (void)out_size; (void)ws_size;
    const float* x      = (const float*)d_in[0];
    const float* ew     = (const float*)d_in[1];
    const float* eb     = (const float*)d_in[2];
    const float* log_dt = (const float*)d_in[3];
    const float* C_re   = (const float*)d_in[4];
    const float* C_im   = (const float*)d_in[5];
    const float* lAr    = (const float*)d_in[6];
    const float* Aim    = (const float*)d_in[7];
    const float* Dp     = (const float*)d_in[8];
    const float* out_w  = (const float*)d_in[9];
    const float* out_b  = (const float*)d_in[10];
    const float* ln_w   = (const float*)d_in[11];
    const float* ln_b   = (const float*)d_in[12];
    const float* dec_w  = (const float*)d_in[13];
    const float* dec_b  = (const float*)d_in[14];

    const size_t nBHL = (size_t)B_ * H_ * L_;
    float*    h_cur  = (float*)d_ws;                              // (B,H,L) f32
    float*    preLN  = h_cur + nBHL;                              // (B*L,H) f32
    _Float16* yf     = (_Float16*)(preLN + nBHL);                 // (B,H,L) f16
    _Float16* w16    = yf + nBHL;                                 // (NL,512,256) f16
    float*    wr     = (float*)(w16 + (size_t)NL_ * O2_ * H_);
    float*    wi     = wr + H_ * N2_;
    float*    crp    = wi + H_ * N2_;
    float*    cip    = crp + H_ * N2_;
    float*    pooled = cip + H_ * N2_;                            // (B,H)

    encode_kernel<<<dim3((unsigned)((nBHL + 255) / 256)), 256, 0, stream>>>(x, ew, eb, h_cur);
    cvt_w_kernel<<<dim3((NL_ * O2_ * H_ + 255) / 256), 256, 0, stream>>>(out_w, w16);

    for (int layer = 0; layer < NL_; ++layer) {
        coeff_kernel<<<dim3((H_ * N2_ + 255) / 256), 256, 0, stream>>>(
            log_dt, C_re, C_im, lAr, Aim, layer, wr, wi, crp, cip);
        ssm_scan_kernel<<<dim3(B_ * H_ / 8), 256, 0, stream>>>(
            h_cur, wr, wi, crp, cip, Dp, layer, yf);
        glu_gemm_kernel<<<dim3((B_ * L_) / 64, H_ / 64), 256, 0, stream>>>(
            yf, w16 + (size_t)layer * O2_ * H_, out_b + layer * O2_, h_cur, preLN);
        layernorm_kernel<<<dim3(B_ * L_), 256, 0, stream>>>(
            preLN, ln_w + layer * H_, ln_b + layer * H_, h_cur);
    }

    pool_kernel<<<dim3(B_ * H_), 256, 0, stream>>>(h_cur, pooled);
    decode_kernel<<<dim3(1), 256, 0, stream>>>(pooled, dec_w, dec_b, (float*)d_out);
}
